// Transformer_6700148982249
// MI455X (gfx1250) — matrix operand order repaired and checked by reading, not compiled, not run on hardware
//
#include <hip/hip_runtime.h>
#include <math.h>

#define USE_TDM 1   // Tensor Data Mover staging in GEMM (inline asm)

typedef __bf16 bf16;
typedef __attribute__((ext_vector_type(16))) __bf16 bf16x16;
typedef __attribute__((ext_vector_type(8)))  __bf16 bf16x8;
typedef __attribute__((ext_vector_type(8)))  float  f32x8;
typedef __attribute__((ext_vector_type(4)))  unsigned int u32x4;
typedef __attribute__((ext_vector_type(8)))  unsigned int u32x8;

#define LAYERS 4
#define NH     12
#define HD     64
#define DMODEL 768
#define DFF    3072
#define NVOCAB 32000
#define SEQ    1024
#define NB     2
#define NTOK   (NB * SEQ)   // 2048

// two contiguous 8xbf16 chunks -> one 16-element fragment
#define LDFRAG(P0, P1)                                                        \
  __builtin_shufflevector(*(const bf16x8*)(P0), *(const bf16x8*)(P1),         \
                          0, 1, 2, 3, 4, 5, 6, 7, 8, 9, 10, 11, 12, 13, 14, 15)

#if USE_TDM
// ---------------------------------------------------------------------------
// TDM: DMA one 2D tile (rows x 32 bf16, row stride = strideElems) from global
// into contiguous LDS. Descriptor packed per CDNA5 ISA sec 8.3/8.4.
// Issued per-wave (EXEC ignored); caller guards to one wave and barriers.
// ---------------------------------------------------------------------------
__device__ __forceinline__ void tdm_load_tile(unsigned ldsAddr, const void* gptr,
                                              unsigned rows, unsigned strideElems) {
  unsigned long long ga = (unsigned long long)(size_t)gptr;
  u32x4 g0;
  g0[0] = 1u;                                               // count=1 (valid), load
  g0[1] = ldsAddr;                                          // lds_addr (bytes)
  g0[2] = (unsigned)ga;                                     // global_addr[31:0]
  g0[3] = (unsigned)((ga >> 32) & 0x1FFFFFFu) | (2u << 30); // addr[56:32] | type=2
  u32x8 g1;
  g1[0] = (1u << 16);               // data_size = 1 (2 bytes)
  g1[1] = (32u << 16);              // tensor_dim0 = 32  (bits 79:48 lo half)
  g1[2] = (rows << 16);             // dim0 hi = 0 ; tensor_dim1 = rows
  g1[3] = (32u << 16);              // dim1 hi = 0 ; tile_dim0 = 32
  g1[4] = rows;                     // tile_dim1 = rows ; tile_dim2 = 0
  g1[5] = strideElems;              // tensor_dim0_stride[31:0] (elements)
  g1[6] = 0u;                       // stride0 hi ; stride1 lo
  g1[7] = 0u;
  u32x4 gz = {0u, 0u, 0u, 0u};      // groups 2/3: unused (2D tile)
  asm volatile("tensor_load_to_lds %0, %1, %2, %3"
               :: "s"(g0), "s"(g1), "s"(gz), "s"(gz) : "memory");
}
#endif

// ---------------------------------------------------------------------------
// Embedding: res[b,s,:] = E_w[x[b,s],:] + P_w[s,:]
// ---------------------------------------------------------------------------
__global__ void embed_kernel(const int* __restrict__ x, const float* __restrict__ E,
                             const float* __restrict__ P, float* __restrict__ res) {
  int row = blockIdx.x;
  int s   = row % SEQ;
  int tok = x[row];
  const float* e = E + (size_t)tok * DMODEL;
  const float* p = P + (size_t)s   * DMODEL;
  float* r = res + (size_t)row * DMODEL;
  for (int i = threadIdx.x; i < DMODEL; i += blockDim.x) r[i] = e[i] + p[i];
}

// ---------------------------------------------------------------------------
// LayerNorm (unbiased std, ddof=1), bf16 output
// ---------------------------------------------------------------------------
__global__ __launch_bounds__(256)
void ln_kernel(const float* __restrict__ x, const float* __restrict__ w,
               const float* __restrict__ b, bf16* __restrict__ out, int D_) {
  int row = blockIdx.x;
  const float* xr = x + (size_t)row * D_;
  __shared__ float ssum[256], ssq[256];
  int tid = threadIdx.x;
  float s = 0.f, q = 0.f;
  for (int i = tid; i < D_; i += 256) { float v = xr[i]; s += v; q += v * v; }
  ssum[tid] = s; ssq[tid] = q; __syncthreads();
  for (int off = 128; off > 0; off >>= 1) {
    if (tid < off) { ssum[tid] += ssum[tid + off]; ssq[tid] += ssq[tid + off]; }
    __syncthreads();
  }
  float mu  = ssum[0] / D_;
  float var = (ssq[0] - D_ * mu * mu) / (D_ - 1);
  float inv = rsqrtf(fmaxf(var, 1e-12f));
  bf16* outr = out + (size_t)row * D_;
  for (int i = tid; i < D_; i += 256)
    outr[i] = (bf16)(w[i] * ((xr[i] - mu) * inv) + b[i]);
}

// ---------------------------------------------------------------------------
// Tiled transpose + cast: in fp32 [R][C]  ->  out bf16 [C][R]
// ---------------------------------------------------------------------------
__global__ __launch_bounds__(256)
void transcast_kernel(const float* __restrict__ in, bf16* __restrict__ out,
                      int R, int C) {
  __shared__ float tile[32][33];
  int r0 = blockIdx.y * 32, c0 = blockIdx.x * 32;
  int tx = threadIdx.x & 31, ty = threadIdx.x >> 5;
  for (int i = ty; i < 32; i += 8)
    tile[i][tx] = in[(size_t)(r0 + i) * C + (c0 + tx)];
  __syncthreads();
  for (int i = ty; i < 32; i += 8)
    out[(size_t)(c0 + i) * R + (r0 + tx)] = (bf16)tile[tx][i];
}

// Q/K/V weight permute+cast: in [H, D, DH] fp32 -> out [H*DH][D] bf16 (B^T)
__global__ void permqkv_kernel(const float* __restrict__ in, bf16* __restrict__ out) {
  int idx = blockIdx.x * 256 + threadIdx.x;
  if (idx < DMODEL * DMODEL) {
    int c = idx / DMODEL;                          // c = h*64+e
    int d = idx % DMODEL;
    int h = c / HD, e = c % HD;
    out[idx] = (bf16)in[((size_t)h * DMODEL + d) * HD + e];
  }
}

__global__ void gelu_kernel(const float* __restrict__ in, bf16* __restrict__ out, int n) {
  int i = blockIdx.x * 256 + threadIdx.x;
  if (i < n) {
    float x = in[i];
    out[i] = (bf16)(0.5f * x * (1.f + erff(x * 0.70710678118654752f)));
  }
}

__global__ void add2_kernel(const float* __restrict__ a, const float* __restrict__ b,
                            float* __restrict__ c, int n) {
  int i = blockIdx.x * 256 + threadIdx.x;
  if (i < n) c[i] = a[i] + b[i];
}

__global__ void add3_kernel(float* __restrict__ res, const float* __restrict__ ra,
                            const float* __restrict__ m, int n) {
  int i = blockIdx.x * 256 + threadIdx.x;
  if (i < n) res[i] = res[i] + ra[i] + m[i];
}

// ---------------------------------------------------------------------------
// WMMA GEMM: C[M,N] = A[M,K](bf16) x B[K,N] + bias[N], B given TRANSPOSED as
// Bt[N][K] bf16. Block = 128 threads (4 waves), 64x64 tile, K-step 32.
// Tiles staged by the Tensor Data Mover (wave 0 issues, TENSORcnt wait).
// outMode: 0 = f32, 1 = bf16, 2 = bf16 transposed for attention V
// ---------------------------------------------------------------------------
__global__ __launch_bounds__(128)
void gemm_kernel(const bf16* __restrict__ A, const bf16* __restrict__ Bt,
                 const float* __restrict__ bias, void* __restrict__ C,
                 int M_, int N_, int K_, int outMode) {
  __shared__ __align__(16) bf16 As[64][32];   // contiguous: TDM-friendly
  __shared__ __align__(16) bf16 Bs[64][32];   // Bs[col][K]
  int tid  = threadIdx.x;
  int wave = tid >> 5, lane = tid & 31, half = lane >> 4, ln = lane & 15;
  int rowBase = blockIdx.y * 64, colBase = blockIdx.x * 64;
  f32x8 acc[4] = {};
  const bf16* aTile = A  + (size_t)rowBase * K_;
  const bf16* bTile = Bt + (size_t)colBase * K_;
#if USE_TDM
  unsigned ldsA = (unsigned)(size_t)(&As[0][0]);
  unsigned ldsB = (unsigned)(size_t)(&Bs[0][0]);
#endif
  for (int k0 = 0; k0 < K_; k0 += 32) {
#if USE_TDM
    if (wave == 0) {
      tdm_load_tile(ldsA, aTile + k0, 64u, (unsigned)K_);
      tdm_load_tile(ldsB, bTile + k0, 64u, (unsigned)K_);
      __builtin_amdgcn_s_wait_tensorcnt(0);
    }
    __syncthreads();
#else
    {   // manual staging: 2 chunks per array per thread, fully unrolled
      int r = tid >> 2, c8 = (tid & 3) << 3;
      float4 va0 = *(const float4*)&aTile[(size_t)r        * K_ + k0 + c8];
      float4 va1 = *(const float4*)&aTile[(size_t)(r + 32) * K_ + k0 + c8];
      float4 vb0 = *(const float4*)&bTile[(size_t)r        * K_ + k0 + c8];
      float4 vb1 = *(const float4*)&bTile[(size_t)(r + 32) * K_ + k0 + c8];
      *(float4*)&As[r][c8]      = va0;
      *(float4*)&As[r + 32][c8] = va1;
      *(float4*)&Bs[r][c8]      = vb0;
      *(float4*)&Bs[r + 32][c8] = vb1;
    }
    __syncthreads();
#endif
    // A fragment, then ALL B fragments before the WMMA burst
    const bf16* ar = &As[wave * 16 + ln][0];
    bf16x16 af = LDFRAG(ar + half * 16, ar + half * 16 + 8);
    bf16x16 b0 = LDFRAG(&Bs[ 0 + ln][half * 16], &Bs[ 0 + ln][half * 16 + 8]);
    bf16x16 b1 = LDFRAG(&Bs[16 + ln][half * 16], &Bs[16 + ln][half * 16 + 8]);
    bf16x16 b2 = LDFRAG(&Bs[32 + ln][half * 16], &Bs[32 + ln][half * 16 + 8]);
    bf16x16 b3 = LDFRAG(&Bs[48 + ln][half * 16], &Bs[48 + ln][half * 16 + 8]);
    acc[0] = __builtin_amdgcn_wmma_f32_16x16x32_bf16(false, af, false, b0, (short)0, acc[0], false, false);
    acc[1] = __builtin_amdgcn_wmma_f32_16x16x32_bf16(false, af, false, b1, (short)0, acc[1], false, false);
    acc[2] = __builtin_amdgcn_wmma_f32_16x16x32_bf16(false, af, false, b2, (short)0, acc[2], false, false);
    acc[3] = __builtin_amdgcn_wmma_f32_16x16x32_bf16(false, af, false, b3, (short)0, acc[3], false, false);
    __syncthreads();
  }
#pragma unroll
  for (int nt = 0; nt < 4; nt++) {
#pragma unroll
    for (int i = 0; i < 8; i++) {
      int row = rowBase + wave * 16 + i + 8 * half;
      int col = colBase + nt * 16 + ln;
      float vv = acc[nt][i] + bias[col];
      if (outMode == 0) {
        ((float*)C)[(size_t)row * N_ + col] = vv;
      } else if (outMode == 1) {
        ((bf16*)C)[(size_t)row * N_ + col] = (bf16)vv;
      } else {
        int bb = row / SEQ, s = row % SEQ;
        int hh = col / HD,  e = col % HD;
        ((bf16*)C)[((size_t)((bb * NH + hh) * HD) + e) * SEQ + s] = (bf16)vv;
      }
    }
  }
}

// ---------------------------------------------------------------------------
// Fused causal attention, online softmax. One wave per (b, h, 16-row q-tile).
// q,k: bf16 [B*S, H*HD]; vt: bf16 [B, H, HD, S] (pre-transposed);
// out: bf16 [B*S, H*HD]. All WMMA fragments are 16-byte loads.
// ---------------------------------------------------------------------------
__global__ __launch_bounds__(32)
void attn_kernel(const bf16* __restrict__ q, const bf16* __restrict__ k,
                 const bf16* __restrict__ vt, bf16* __restrict__ out) {
  const int ld = NH * HD;
  int bid = blockIdx.x;
  int qt  = bid % (SEQ / 16);
  int h   = (bid / (SEQ / 16)) % NH;
  int b   = bid / ((SEQ / 16) * NH);
  int lane = threadIdx.x & 31;
  int half = lane >> 4, ln = lane & 15;
  int qbase = qt * 16;

  const bf16* qp = q + (size_t)(b * SEQ + qbase + ln) * ld + h * HD;
  bf16x16 aq0 = LDFRAG(qp + half * 16,      qp + half * 16 + 8);
  bf16x16 aq1 = LDFRAG(qp + 32 + half * 16, qp + 32 + half * 16 + 8);

  float mrow[8], srow[8];
  f32x8 oacc[4] = {};
#pragma unroll
  for (int i = 0; i < 8; i++) { mrow[i] = -3.0e38f; srow[i] = 0.f; }
  const float scale = 0.125f;
  __shared__ __align__(16) bf16 pl[16][40];

  for (int kk = 0; kk <= qbase; kk += 32) {
    f32x8 cl[2] = {};
#pragma unroll
    for (int nt = 0; nt < 2; nt++) {
      int krow = kk + nt * 16 + ln;
      if (krow > SEQ - 1) krow = SEQ - 1;
      const bf16* kp = k + (size_t)(b * SEQ + krow) * ld + h * HD;
      bf16x16 b0 = LDFRAG(kp + half * 16,      kp + half * 16 + 8);
      bf16x16 b1 = LDFRAG(kp + 32 + half * 16, kp + 32 + half * 16 + 8);
      cl[nt] = __builtin_amdgcn_wmma_f32_16x16x32_bf16(
          false, aq0, false, b0, (short)0, cl[nt], false, false);
      cl[nt] = __builtin_amdgcn_wmma_f32_16x16x32_bf16(
          false, aq1, false, b1, (short)0, cl[nt], false, false);
    }
#pragma unroll
    for (int nt = 0; nt < 2; nt++) {
#pragma unroll
      for (int i = 0; i < 8; i++) {
        int grow = qbase + i + 8 * half;
        int gcol = kk + nt * 16 + ln;
        float vv = cl[nt][i] * scale;
        cl[nt][i] = (gcol <= grow) ? vv : -3.0e38f;
      }
    }
#pragma unroll
    for (int i = 0; i < 8; i++) {
      float rm = fmaxf(cl[0][i], cl[1][i]);
#pragma unroll
      for (int d = 1; d < 16; d <<= 1) rm = fmaxf(rm, __shfl_xor(rm, d, 32));
      float nm   = fmaxf(mrow[i], rm);
      float corr = __expf(mrow[i] - nm);
      float p0   = __expf(cl[0][i] - nm);
      float p1   = __expf(cl[1][i] - nm);
      float rs   = p0 + p1;
#pragma unroll
      for (int d = 1; d < 16; d <<= 1) rs += __shfl_xor(rs, d, 32);
      srow[i] = srow[i] * corr + rs;
      mrow[i] = nm;
#pragma unroll
      for (int nt = 0; nt < 4; nt++) oacc[nt][i] *= corr;
      pl[i + 8 * half][ln]      = (bf16)p0;
      pl[i + 8 * half][16 + ln] = (bf16)p1;
    }
    __syncthreads();
    const bf16* pr = &pl[ln][0];
    bf16x16 ap = LDFRAG(pr + half * 16, pr + half * 16 + 8);
#pragma unroll
    for (int nt = 0; nt < 4; nt++) {
      const bf16* vp = vt + ((size_t)((b * NH + h) * HD) + nt * 16 + ln) * SEQ + kk;
      bf16x16 bv = LDFRAG(vp + half * 16, vp + half * 16 + 8);
      oacc[nt] = __builtin_amdgcn_wmma_f32_16x16x32_bf16(
          false, ap, false, bv, (short)0, oacc[nt], false, false);
    }
    __syncthreads();
  }
#pragma unroll
  for (int nt = 0; nt < 4; nt++) {
#pragma unroll
    for (int i = 0; i < 8; i++) {
      int grow = qbase + i + 8 * half;
      float vv = oacc[nt][i] / srow[i];
      out[(size_t)(b * SEQ + grow) * ld + h * HD + nt * 16 + ln] = (bf16)vv;
    }
  }
}

// ---------------------------------------------------------------------------
// Host orchestration
// ---------------------------------------------------------------------------
extern "C" void kernel_launch(void* const* d_in, const int* in_sizes, int n_in,
                              void* d_out, int out_size, void* d_ws, size_t ws_size,
                              hipStream_t stream) {
  (void)in_sizes; (void)n_in; (void)out_size; (void)ws_size;
  const int*   x    = (const int*)  d_in[0];
  const float* E_w  = (const float*)d_in[1];
  const float* P_w  = (const float*)d_in[2];
  const float* ln1w = (const float*)d_in[3];
  const float* ln1b = (const float*)d_in[4];
  const float* Qw   = (const float*)d_in[5];
  const float* Qb   = (const float*)d_in[6];
  const float* Kw   = (const float*)d_in[7];
  const float* Kb   = (const float*)d_in[8];
  const float* Vw   = (const float*)d_in[9];
  const float* Vb   = (const float*)d_in[10];
  const float* Ow   = (const float*)d_in[11];
  const float* Ob   = (const float*)d_in[12];
  const float* ln2w = (const float*)d_in[13];
  const float* ln2b = (const float*)d_in[14];
  const float* Inw  = (const float*)d_in[15];
  const float* Inb  = (const float*)d_in[16];
  const float* Outw = (const float*)d_in[17];
  const float* Outb = (const float*)d_in[18];
  const float* lnfw = (const float*)d_in[19];
  const float* lnfb = (const float*)d_in[20];
  const float* Uw   = (const float*)d_in[21];
  const float* Ub   = (const float*)d_in[22];

  char* ws = (char*)d_ws;
  size_t off = 0;
  auto wsalloc = [&](size_t bytes) -> char* {
    char* p = ws + off;
    off += (bytes + 255) & ~(size_t)255;
    return p;
  };
  float* res   = (float*)wsalloc((size_t)NTOK * DMODEL * 4);
  float* resat = (float*)wsalloc((size_t)NTOK * DMODEL * 4);
  bf16*  hbf   = (bf16*) wsalloc((size_t)NTOK * DMODEL * 2);
  bf16*  qbuf  = (bf16*) wsalloc((size_t)NTOK * DMODEL * 2);
  bf16*  kbuf  = (bf16*) wsalloc((size_t)NTOK * DMODEL * 2);
  bf16*  vtbuf = (bf16*) wsalloc((size_t)NTOK * DMODEL * 2 + 256);
  bf16*  abuf  = (bf16*) wsalloc((size_t)NTOK * DMODEL * 2);
  char*  big   = wsalloc((size_t)51904512);
  bf16*  Wq    = (bf16*)(big + 0);            // all weights stored TRANSPOSED [N][K]
  bf16*  Wk    = (bf16*)(big + 1179648);
  bf16*  Wv    = (bf16*)(big + 2359296);
  bf16*  WoO   = (bf16*)(big + 3538944);
  bf16*  Wi    = (bf16*)(big + 4718592);
  bf16*  Wo2   = (bf16*)(big + 9437184);
  bf16*  gbuf  = (bf16*)(big + 14155776);
  float* scrF  = (float*)(big + 26738688);
  bf16*  Ubf   = (bf16*)(big + 0);            // reused after the layer loop

  const int pgrid  = (DMODEL * DMODEL + 255) / 256;
  const int ndgrid = (NTOK * DMODEL + 255) / 256;

  embed_kernel<<<NTOK, 256, 0, stream>>>(x, E_w, P_w, res);

  for (int l = 0; l < LAYERS; l++) {
    ln_kernel<<<NTOK, 256, 0, stream>>>(res, ln1w + l * DMODEL, ln1b + l * DMODEL, hbf, DMODEL);
    permqkv_kernel<<<pgrid, 256, 0, stream>>>(Qw + (size_t)l * NH * DMODEL * HD, Wq);
    permqkv_kernel<<<pgrid, 256, 0, stream>>>(Kw + (size_t)l * NH * DMODEL * HD, Wk);
    permqkv_kernel<<<pgrid, 256, 0, stream>>>(Vw + (size_t)l * NH * DMODEL * HD, Wv);
    transcast_kernel<<<dim3(DMODEL / 32, DMODEL / 32), 256, 0, stream>>>(
        Ow + (size_t)l * NH * HD * DMODEL, WoO, DMODEL, DMODEL);
    dim3 g1(DMODEL / 64, NTOK / 64);
    gemm_kernel<<<g1, 128, 0, stream>>>(hbf, Wq, Qb + l * DMODEL, qbuf, NTOK, DMODEL, DMODEL, 1);
    gemm_kernel<<<g1, 128, 0, stream>>>(hbf, Wk, Kb + l * DMODEL, kbuf, NTOK, DMODEL, DMODEL, 1);
    gemm_kernel<<<g1, 128, 0, stream>>>(hbf, Wv, Vb + l * DMODEL, vtbuf, NTOK, DMODEL, DMODEL, 2);
    attn_kernel<<<NB * NH * (SEQ / 16), 32, 0, stream>>>(qbuf, kbuf, vtbuf, abuf);
    gemm_kernel<<<g1, 128, 0, stream>>>(abuf, WoO, Ob + l * DMODEL, scrF, NTOK, DMODEL, DMODEL, 0);
    add2_kernel<<<ndgrid, 256, 0, stream>>>(res, scrF, resat, NTOK * DMODEL);
    ln_kernel<<<NTOK, 256, 0, stream>>>(resat, ln2w + l * DMODEL, ln2b + l * DMODEL, hbf, DMODEL);
    transcast_kernel<<<dim3(DFF / 32, DMODEL / 32), 256, 0, stream>>>(
        Inw + (size_t)l * DMODEL * DFF, Wi, DMODEL, DFF);
    transcast_kernel<<<dim3(DMODEL / 32, DFF / 32), 256, 0, stream>>>(
        Outw + (size_t)l * DFF * DMODEL, Wo2, DFF, DMODEL);
    dim3 g2(DFF / 64, NTOK / 64);
    gemm_kernel<<<g2, 128, 0, stream>>>(hbf, Wi, Inb + l * DFF, scrF, NTOK, DFF, DMODEL, 0);
    gelu_kernel<<<(NTOK * DFF + 255) / 256, 256, 0, stream>>>(scrF, gbuf, NTOK * DFF);
    dim3 g3(DMODEL / 64, NTOK / 64);
    gemm_kernel<<<g3, 128, 0, stream>>>(gbuf, Wo2, Outb + l * DMODEL, scrF, NTOK, DMODEL, DFF, 0);
    add3_kernel<<<ndgrid, 256, 0, stream>>>(res, resat, scrF, NTOK * DMODEL);
  }

  ln_kernel<<<NTOK, 256, 0, stream>>>(res, lnfw, lnfb, hbf, DMODEL);
  transcast_kernel<<<dim3(NVOCAB / 32, DMODEL / 32), 256, 0, stream>>>(
      Uw, Ubf, DMODEL, NVOCAB);
  dim3 g4(NVOCAB / 64, NTOK / 64);
  gemm_kernel<<<g4, 128, 0, stream>>>(hbf, Ubf, Ub, (float*)d_out, NTOK, NVOCAB, DMODEL, 0);
}
